// LatentCorrelationLayer_7499012899246
// MI455X (gfx1250) — matrix operand order repaired and checked by reading, not compiled, not run on hardware
//
#include <hip/hip_runtime.h>
#include <math.h>

// Problem constants (from reference)
#define B_  4
#define T_  64
#define N_  2048
#define C_  32
#define H_  64
#define G_  192      // 3*H, gate dim (r,z,n)
#define M_  8192     // B*N
#define SLOPE_ 0.2f

typedef __attribute__((ext_vector_type(16))) __bf16 v16bf;
typedef __attribute__((ext_vector_type(8)))  __bf16 v8bf;
typedef __attribute__((ext_vector_type(8)))  float  v8f;

static __device__ __forceinline__ v8f wmma_bf16(v16bf a, v16bf b, v8f c) {
  // D = A(16x32 bf16) * B(32x16 bf16) + C(16x16 f32)
  return __builtin_amdgcn_wmma_f32_16x16x32_bf16(
      /*neg_a=*/false, a, /*neg_b=*/false, b,
      /*c_mod=*/(short)0, c, /*reuse_a=*/false, /*reuse_b=*/false);
}

// Fast activations: one v_exp_f32 + one v_rcp_f32 each (no IEEE div sequence).
static __device__ __forceinline__ float fast_sigmoid(float x) {
  return __builtin_amdgcn_rcpf(1.0f + __expf(-x));
}
static __device__ __forceinline__ float fast_tanh(float x) {
  // tanh(x) = 1 - 2/(exp(2x)+1); saturates to +/-1 at the extremes.
  return 1.0f - 2.0f * __builtin_amdgcn_rcpf(__expf(2.0f * x) + 1.0f);
}

static __device__ __forceinline__ v16bf pack16(v8bf lo, v8bf hi) {
  v16bf r;
#pragma unroll
  for (int e = 0; e < 8; ++e) { r[e] = lo[e]; r[e + 8] = hi[e]; }
  return r;
}

// B-matrix fragment (KxN = 32x16) of W^T, i.e. element = W[ncol, k].
// Lane layout: col = lane&15; e<8 -> k = 8*(lane>>4)+e, e>=8 -> k = 16 + 8*(lane>>4)+(e-8).
static __device__ __forceinline__ v16bf load_bfrag(const __bf16* W, int stride,
                                                   int ncol, int ksub, int lane) {
  const __bf16* p = W + ncol * stride + ksub + ((lane >> 4) << 3);
  v8bf lo = *(const v8bf*)p;
  v8bf hi = *(const v8bf*)(p + 16);
  return pack16(lo, hi);
}

// A-matrix fragment (16x32) of h from per-wave LDS tile (row-major 16x64).
// Lane layout: row = lane&15; e<8 -> k = 8*(lane>>4)+e (+32*f),
//              e>=8 -> k = 16 + 8*(lane>>4)+(e-8) (+32*f).
static __device__ __forceinline__ v16bf load_afrag_h(const __bf16* sHw,
                                                     int lane, int f) {
  int row = lane & 15;
  int kb  = ((lane >> 4) << 3) + (f << 5);
  const __bf16* p = sHw + row * H_ + kb;
  v8bf lo = *(const v8bf*)p;         // k = kb .. kb+7
  v8bf hi = *(const v8bf*)(p + 16);  // k = kb+16 .. kb+23
  return pack16(lo, hi);
}

#define WAVES_ 2
#define BLK_   (WAVES_ * 32)

// ---------------------------------------------------------------------------
// Kernel 1: GRU over all T for a private 16-row M-tile per wave (wave32).
// Weight B-fragments are loop-invariant -> compiler keeps them register-
// resident (CDNA5 1024-VGPR file). Emits key[m]=h_T.wk, query[m]=h_T.wq.
// ---------------------------------------------------------------------------
__global__ __launch_bounds__(BLK_) void gru_wmma_kernel(
    const float* __restrict__ x,     // [B,T,N,C]
    const float* __restrict__ w_ih,  // [3H,C]
    const float* __restrict__ w_hh,  // [3H,H]
    const float* __restrict__ b_ih,  // [3H]
    const float* __restrict__ b_hh,  // [3H]
    const float* __restrict__ wk,    // [H]
    const float* __restrict__ wq,    // [H]
    float* __restrict__ keyO,        // [M]
    float* __restrict__ qryO) {      // [M]
  __shared__ __bf16 sWih[G_ * C_];         // 12 KB
  __shared__ __bf16 sWhh[G_ * H_];         // 24 KB
  __shared__ float  sBih[G_], sBhh[G_];
  __shared__ float  sWk[H_], sWq[H_];
  __shared__ __bf16 sH[WAVES_][16 * H_];   // per-wave h staging (D->A reshape)

  const int tid = threadIdx.x;
  for (int i = tid; i < G_ * C_; i += BLK_) sWih[i] = (__bf16)w_ih[i];
  for (int i = tid; i < G_ * H_; i += BLK_) sWhh[i] = (__bf16)w_hh[i];
  for (int i = tid; i < G_; i += BLK_) { sBih[i] = b_ih[i]; sBhh[i] = b_hh[i]; }
  for (int i = tid; i < H_; i += BLK_) { sWk[i] = wk[i]; sWq[i] = wq[i]; }
  __syncthreads();

  const int wv   = tid >> 5;
  const int lane = tid & 31;
  __bf16* sHw = &sH[wv][0];

  const int m0 = (blockIdx.x * WAVES_ + wv) << 4;  // 16 rows per wave
  const int b  = m0 >> 11;                         // N_ = 2048
  const int n0 = m0 & (N_ - 1);                    // tile stays inside one b
  const float* xb0 = x + ((size_t)b * T_ * N_ + (size_t)n0) * C_;

  // h = 0 (both the LDS staging copy and the D-layout register copy)
  for (int i = lane; i < 16 * H_; i += 32) sHw[i] = (__bf16)0.0f;
  v8f hD[4];
#pragma unroll
  for (int c = 0; c < 4; ++c)
#pragma unroll
    for (int r = 0; r < 8; ++r) hD[c][r] = 0.0f;

  const int row  = lane & 15;
  const int koff = (lane >> 4) << 3;
  const int gcol = lane & 15;

  for (int t = 0; t < T_; ++t) {
    const float* xb = xb0 + (size_t)t * N_ * C_;

    // ---- x A-fragment (16x32), contiguous 32B-aligned float4 loads ----
    const float* px = xb + row * C_ + koff;
    float4 f0 = *(const float4*)(px);
    float4 f1 = *(const float4*)(px + 4);
    float4 f2 = *(const float4*)(px + 16);
    float4 f3 = *(const float4*)(px + 20);
    v16bf ax;
    ax[0]  = (__bf16)f0.x; ax[1]  = (__bf16)f0.y; ax[2]  = (__bf16)f0.z; ax[3]  = (__bf16)f0.w;
    ax[4]  = (__bf16)f1.x; ax[5]  = (__bf16)f1.y; ax[6]  = (__bf16)f1.z; ax[7]  = (__bf16)f1.w;
    ax[8]  = (__bf16)f2.x; ax[9]  = (__bf16)f2.y; ax[10] = (__bf16)f2.z; ax[11] = (__bf16)f2.w;
    ax[12] = (__bf16)f3.x; ax[13] = (__bf16)f3.y; ax[14] = (__bf16)f3.z; ax[15] = (__bf16)f3.w;
    if (t + 1 < T_) __builtin_prefetch(px + (size_t)N_ * C_, 0, 1);

    // ---- h A-fragments (16x64 -> two 16x32 K-tiles) from LDS ----
    v16bf ah0 = load_afrag_h(sHw, lane, 0);
    v16bf ah1 = load_afrag_h(sHw, lane, 1);

#pragma unroll
    for (int c = 0; c < 4; ++c) {                 // 4 column tiles of H
      const int nR = c * 16 + gcol;               // r-gate output column
      const int nZ = nR + 64;                     // z-gate
      const int nN = nR + 128;                    // n-gate

      v8f accR, accZ, accIN, accHN;
      const float bR  = sBih[nR] + sBhh[nR];
      const float bZ  = sBih[nZ] + sBhh[nZ];
      const float bIN = sBih[nN];
      const float bHN = sBhh[nN];
#pragma unroll
      for (int r = 0; r < 8; ++r) {
        accR[r] = bR; accZ[r] = bZ; accIN[r] = bIN; accHN[r] = bHN;
      }

      // r = sigmoid(x Wih_r + h Whh_r + b)
      accR  = wmma_bf16(ax,  load_bfrag(sWih, C_, nR, 0,  lane), accR);
      accR  = wmma_bf16(ah0, load_bfrag(sWhh, H_, nR, 0,  lane), accR);
      accR  = wmma_bf16(ah1, load_bfrag(sWhh, H_, nR, 32, lane), accR);
      // z likewise
      accZ  = wmma_bf16(ax,  load_bfrag(sWih, C_, nZ, 0,  lane), accZ);
      accZ  = wmma_bf16(ah0, load_bfrag(sWhh, H_, nZ, 0,  lane), accZ);
      accZ  = wmma_bf16(ah1, load_bfrag(sWhh, H_, nZ, 32, lane), accZ);
      // n-gate: i_n and h_n must stay separate (r scales only h_n)
      accIN = wmma_bf16(ax,  load_bfrag(sWih, C_, nN, 0,  lane), accIN);
      accHN = wmma_bf16(ah0, load_bfrag(sWhh, H_, nN, 0,  lane), accHN);
      accHN = wmma_bf16(ah1, load_bfrag(sWhh, H_, nN, 32, lane), accHN);

#pragma unroll
      for (int r = 0; r < 8; ++r) {
        float rv = fast_sigmoid(accR[r]);
        float zv = fast_sigmoid(accZ[r]);
        float nv = fast_tanh(accIN[r] + rv * accHN[r]);
        hD[c][r] = (1.0f - zv) * nv + zv * hD[c][r];
      }
      // scatter h_new back to the per-wave LDS tile (D-layout -> row-major)
#pragma unroll
      for (int r = 0; r < 8; ++r) {
        sHw[(r + ((lane >> 4) << 3)) * H_ + c * 16 + gcol] = (__bf16)hD[c][r];
      }
    }
  }

  // key/query projections of the final hidden state (one lane per row)
  if (lane < 16) {
    float kk = 0.0f, qq = 0.0f;
#pragma unroll 8
    for (int j = 0; j < H_; ++j) {
      float hv = (float)sHw[lane * H_ + j];
      kk += hv * sWk[j];
      qq += hv * sWq[j];
    }
    keyO[m0 + lane] = kk;
    qryO[m0 + lane] = qq;
  }
}

// ---------------------------------------------------------------------------
// Kernel 2: per-row leaky-ReLU + softmax + batch-mean.
// scores[b,i,j] = key[b,i] + query[b,j] (rank-2, no GEMM needed).
// One block per output row i; all 4 batches' query rows staged in LDS.
// ---------------------------------------------------------------------------
__global__ __launch_bounds__(256) void attn_softmax_mean_kernel(
    const float* __restrict__ keyI, const float* __restrict__ qryI,
    float* __restrict__ out) {
  __shared__ float sQ[B_ * N_];   // 32 KB
  __shared__ float sred[256];
  const int i = blockIdx.x, tid = threadIdx.x;
  for (int idx = tid; idx < B_ * N_; idx += 256) sQ[idx] = qryI[idx];
  __syncthreads();

  float acc[8];
#pragma unroll
  for (int jj = 0; jj < 8; ++jj) acc[jj] = 0.0f;

  for (int b = 0; b < B_; ++b) {
    const float kb = keyI[b * N_ + i];
    float v[8], mx = -3.402823466e38f;
#pragma unroll
    for (int jj = 0; jj < 8; ++jj) {
      const int j = jj * 256 + tid;
      float s = kb + sQ[b * N_ + j];
      s = (s >= 0.0f) ? s : SLOPE_ * s;
      v[jj] = s;
      mx = fmaxf(mx, s);
    }
    sred[tid] = mx; __syncthreads();
    for (int off = 128; off > 0; off >>= 1) {
      if (tid < off) sred[tid] = fmaxf(sred[tid], sred[tid + off]);
      __syncthreads();
    }
    mx = sred[0]; __syncthreads();

    float se = 0.0f;
#pragma unroll
    for (int jj = 0; jj < 8; ++jj) { float e = __expf(v[jj] - mx); v[jj] = e; se += e; }
    sred[tid] = se; __syncthreads();
    for (int off = 128; off > 0; off >>= 1) {
      if (tid < off) sred[tid] += sred[tid + off];
      __syncthreads();
    }
    const float inv = 0.25f * __builtin_amdgcn_rcpf(sred[0]);  // fused mean over B=4
    __syncthreads();
#pragma unroll
    for (int jj = 0; jj < 8; ++jj) acc[jj] += v[jj] * inv;
  }

  const size_t base = (size_t)i * N_;
#pragma unroll
  for (int jj = 0; jj < 8; ++jj) out[base + jj * 256 + tid] = acc[jj];
}

// ---------------------------------------------------------------------------
// Kernel 3: in-place symmetrization; each unordered pair (i<j) is owned by
// exactly one thread -> race-free under graph replay.
// ---------------------------------------------------------------------------
__global__ __launch_bounds__(256) void symmetrize_kernel(float* __restrict__ out) {
  const int j = blockIdx.x * 16 + threadIdx.x;
  const int i = blockIdx.y * 16 + threadIdx.y;
  if (j <= i) return;
  const size_t ij = (size_t)i * N_ + j;
  const size_t ji = (size_t)j * N_ + i;
  const float s = 0.5f * (out[ij] + out[ji]);
  out[ij] = s;
  out[ji] = s;
}

extern "C" void kernel_launch(void* const* d_in, const int* in_sizes, int n_in,
                              void* d_out, int out_size, void* d_ws, size_t ws_size,
                              hipStream_t stream) {
  (void)in_sizes; (void)n_in; (void)out_size; (void)ws_size;
  const float* x    = (const float*)d_in[0];
  const float* w_ih = (const float*)d_in[1];
  const float* w_hh = (const float*)d_in[2];
  const float* b_ih = (const float*)d_in[3];
  const float* b_hh = (const float*)d_in[4];
  const float* wk   = (const float*)d_in[5];
  const float* wq   = (const float*)d_in[6];
  float* out = (float*)d_out;

  float* keyW = (float*)d_ws;        // [M_]
  float* qryW = keyW + M_;           // [M_]  (64 KB total workspace)

  gru_wmma_kernel<<<M_ / (WAVES_ * 16), BLK_, 0, stream>>>(
      x, w_ih, w_hh, b_ih, b_hh, wk, wq, keyW, qryW);

  attn_softmax_mean_kernel<<<N_, 256, 0, stream>>>(keyW, qryW, out);

  dim3 g3(N_ / 16, N_ / 16), b3(16, 16);
  symmetrize_kernel<<<g3, b3, 0, stream>>>(out);
}
